// SGATLayer_188978561164
// MI455X (gfx1250) — compile-verified, matrix operand order checked by reading, not run on hardware
//
#include <hip/hip_runtime.h>

#define N_NODESC 50000
#define N_EDGESC 640000
#define IN_FC 128
#define OUT_FC 32
#define HEADSC 4
#define LRELU_ALPHA 0.2f

typedef __attribute__((ext_vector_type(16))) __bf16 v16bf;
typedef __attribute__((ext_vector_type(8)))  float  v8f;

union BF16x16 { v16bf v; unsigned int u[8]; };

// ---- workspace layout (in floats) ----
#define WS_WH    ((size_t)0)                      // 50000*128   = 6,400,000
#define WS_ASRC  ((size_t)6400000)                // 50000*4     =   200,000
#define WS_ADST  ((size_t)6600000)                // 50000*4     =   200,000
#define WS_WEA   ((size_t)6800000)                // 128*4       =       512
#define WS_SCORE ((size_t)6800512)                // 640000*4    = 2,560,000
#define WS_RED   ((size_t)9360512)                // gmax[4] (uint) + gsum[4]

__device__ __forceinline__ unsigned int f32_to_bf16(float x) {
  unsigned int u = __float_as_uint(x);
  return (u + 0x7FFFu + ((u >> 16) & 1u)) >> 16;   // RNE truncate to bf16 bits
}
__device__ __forceinline__ float bf16_to_f32(unsigned int b) {
  return __uint_as_float(b << 16);
}
// monotone float<->uint encoding for atomicMax over signed floats
__device__ __forceinline__ unsigned int fenc(float f) {
  unsigned int u = __float_as_uint(f);
  return (u & 0x80000000u) ? ~u : (u | 0x80000000u);
}
__device__ __forceinline__ float fdec(unsigned int k) {
  return __uint_as_float((k & 0x80000000u) ? (k & 0x7FFFFFFFu) : ~k);
}

// ---------------------------------------------------------------------------
// 0) init: zero output accumulator, init softmax max/sum reduction slots
// ---------------------------------------------------------------------------
__global__ void k_init(float* __restrict__ out, float* __restrict__ ws) {
  size_t i = (size_t)blockIdx.x * blockDim.x + threadIdx.x;
  float4* o4 = (float4*)out;
  const size_t n4 = (size_t)N_NODESC * IN_FC / 4;
  for (size_t k = i; k < n4; k += (size_t)gridDim.x * blockDim.x)
    o4[k] = make_float4(0.f, 0.f, 0.f, 0.f);
  if (i < 4) {
    ((unsigned int*)(ws + WS_RED))[i] = 0u;       // fenc(-inf) < any real key
    ws[WS_RED + 4 + i] = 0.f;
  }
}

// ---------------------------------------------------------------------------
// 1) w_ea[k][h] = sum_n W_e[k][h*32+n] * a_edge[h][n]   (tiny: 128x4)
// ---------------------------------------------------------------------------
__global__ void k_wea(const float* __restrict__ We, const float* __restrict__ a_edge,
                      float* __restrict__ ws) {
  int k = threadIdx.x;                            // 128 threads
  float* wea = ws + WS_WEA;
  for (int h = 0; h < HEADSC; ++h) {
    float s = 0.f;
    #pragma unroll 8
    for (int n = 0; n < OUT_FC; ++n)
      s += We[k * 128 + h * 32 + n] * a_edge[h * 32 + n];
    wea[k * 4 + h] = s;
  }
}

// ---------------------------------------------------------------------------
// 2) Wh = h @ W via WMMA bf16 3-term split (hi*hi + hi*lo + lo*hi), f32 acc.
//    Block: 256 thr = 8 waves; wave = one 16-row M tile x full N=128.
//    W pre-split & pre-swizzled into LDS in exact WMMA B-fragment layout.
// ---------------------------------------------------------------------------
__global__ void __launch_bounds__(256)
k_gemm_wh(const float* __restrict__ H, const float* __restrict__ W,
          float* __restrict__ ws) {
  __shared__ unsigned int ldsW[2][4][8][32][8];   // [hi/lo][kt][nt][lane][pair] = 64 KB

  const int tid = threadIdx.x;
  // ---- cooperative stage of W (hi/lo bf16, fragment layout) ----
  for (int idx = tid; idx < 8192; idx += 256) {
    int p  = idx & 7;
    int l  = (idx >> 3) & 31;
    int nt = (idx >> 8) & 7;
    int kt = (idx >> 11) & 3;
    int j0 = p * 2;
    // element j of B-frag in lane l holds K = kt*32 + koff(l,j), N = nt*16 + (l&15)
    int K  = kt * 32 + (j0 & 7) + ((j0 & 8) << 1) + ((l & 16) >> 1);
    int Nc = nt * 16 + (l & 15);
    float w0 = W[K * 128 + Nc];
    float w1 = W[(K + 1) * 128 + Nc];
    unsigned int h0 = f32_to_bf16(w0), h1 = f32_to_bf16(w1);
    ldsW[0][kt][nt][l][p] = h0 | (h1 << 16);
    unsigned int l0 = f32_to_bf16(w0 - bf16_to_f32(h0));
    unsigned int l1 = f32_to_bf16(w1 - bf16_to_f32(h1));
    ldsW[1][kt][nt][l][p] = l0 | (l1 << 16);
  }
  __syncthreads();

  const int tile = blockIdx.x * 8 + (tid >> 5);   // 3125 tiles of 16 rows (exact)
  if (tile >= (N_NODESC / 16)) return;
  const int lane = tid & 31;
  const int half = lane >> 4;                     // K-half select per ISA A layout
  const int row0 = tile * 16;
  const int arow = row0 + (lane & 15);

  // ---- load + split A fragments (4 K-chunks of 32) ----
  BF16x16 aHi[4], aLo[4];
  const float* hrow = H + (size_t)arow * IN_FC;
  #pragma unroll
  for (int kt = 0; kt < 4; ++kt) {
    int base = kt * 32 + half * 8;
    float4 f0 = *(const float4*)(hrow + base);
    float4 f1 = *(const float4*)(hrow + base + 4);
    float4 f2 = *(const float4*)(hrow + base + 16);
    float4 f3 = *(const float4*)(hrow + base + 20);
    float vals[16] = {f0.x, f0.y, f0.z, f0.w, f1.x, f1.y, f1.z, f1.w,
                      f2.x, f2.y, f2.z, f2.w, f3.x, f3.y, f3.z, f3.w};
    #pragma unroll
    for (int p = 0; p < 8; ++p) {
      float x0 = vals[2 * p], x1 = vals[2 * p + 1];
      unsigned int h0 = f32_to_bf16(x0), h1 = f32_to_bf16(x1);
      aHi[kt].u[p] = h0 | (h1 << 16);
      unsigned int r0 = f32_to_bf16(x0 - bf16_to_f32(h0));
      unsigned int r1 = f32_to_bf16(x1 - bf16_to_f32(h1));
      aLo[kt].u[p] = r0 | (r1 << 16);
    }
  }

  float* Wh = ws + WS_WH;
  #pragma unroll
  for (int nt = 0; nt < 8; ++nt) {
    v8f acc = {0.f, 0.f, 0.f, 0.f, 0.f, 0.f, 0.f, 0.f};
    #pragma unroll
    for (int kt = 0; kt < 4; ++kt) {
      BF16x16 bh, bl;
      #pragma unroll
      for (int p = 0; p < 8; ++p) {
        bh.u[p] = ldsW[0][kt][nt][lane][p];
        bl.u[p] = ldsW[1][kt][nt][lane][p];
      }
      acc = __builtin_amdgcn_wmma_f32_16x16x32_bf16(false, aHi[kt].v, false, bh.v,
                                                    (short)0, acc, false, false);
      acc = __builtin_amdgcn_wmma_f32_16x16x32_bf16(false, aHi[kt].v, false, bl.v,
                                                    (short)0, acc, false, false);
      acc = __builtin_amdgcn_wmma_f32_16x16x32_bf16(false, aLo[kt].v, false, bh.v,
                                                    (short)0, acc, false, false);
    }
    // D layout: lane holds col nt*16+(lane&15), rows row0 + half*8 + v
    int col = nt * 16 + (lane & 15);
    #pragma unroll
    for (int v = 0; v < 8; ++v)
      Wh[(size_t)(row0 + half * 8 + v) * IN_FC + col] = acc[v];
  }
}

// ---------------------------------------------------------------------------
// 3) per-node attention dots: alpha_src[n][h], alpha_dst[n][h]  (wave per node)
// ---------------------------------------------------------------------------
__global__ void k_alpha_node(const float* __restrict__ a_src,
                             const float* __restrict__ a_dst,
                             float* __restrict__ ws) {
  int node = blockIdx.x * 8 + (threadIdx.x >> 5);
  int lane = threadIdx.x & 31;
  int head = lane >> 3, sub = lane & 7;
  const float4 v  = ((const float4*)(ws + WS_WH + (size_t)node * IN_FC))[lane];
  const float4 as = ((const float4*)a_src)[head * 8 + sub];
  const float4 ad = ((const float4*)a_dst)[head * 8 + sub];
  float ps = v.x * as.x + v.y * as.y + v.z * as.z + v.w * as.w;
  float pd = v.x * ad.x + v.y * ad.y + v.z * ad.z + v.w * ad.w;
  #pragma unroll
  for (int off = 1; off < 8; off <<= 1) {
    ps += __shfl_xor(ps, off);
    pd += __shfl_xor(pd, off);
  }
  if (sub == 0) {
    ws[WS_ASRC + (size_t)node * 4 + head] = ps;
    ws[WS_ADST + (size_t)node * 4 + head] = pd;
  }
}

// ---------------------------------------------------------------------------
// 4) edge scores: alpha_e = edge_feat @ w_ea, leaky-relu, global per-head max
//    (wave per edge; streams 327 MB of edge_feat — the HBM-bound pass)
// ---------------------------------------------------------------------------
__global__ void __launch_bounds__(256)
k_edge_score(const int* __restrict__ ei, const float* __restrict__ ef,
             float* __restrict__ ws) {
  __shared__ unsigned int smax[4];
  int tid = threadIdx.x;
  if (tid < 4) smax[tid] = 0u;
  __syncthreads();

  int edge = blockIdx.x * 8 + (tid >> 5);
  int lane = tid & 31;
  const float* wea = ws + WS_WEA;

  float4 v = ((const float4*)(ef + (size_t)edge * IN_FC))[lane];
  float4 part = make_float4(0.f, 0.f, 0.f, 0.f);
  #pragma unroll
  for (int j = 0; j < 4; ++j) {
    float vj = (j == 0) ? v.x : (j == 1) ? v.y : (j == 2) ? v.z : v.w;
    float4 w4 = ((const float4*)wea)[4 * lane + j];   // rows 4*lane+j, all 4 heads
    part.x += vj * w4.x; part.y += vj * w4.y;
    part.z += vj * w4.z; part.w += vj * w4.w;
  }
  #pragma unroll
  for (int off = 16; off > 0; off >>= 1) {
    part.x += __shfl_xor(part.x, off);
    part.y += __shfl_xor(part.y, off);
    part.z += __shfl_xor(part.z, off);
    part.w += __shfl_xor(part.w, off);
  }
  int src = ei[edge], dst = ei[N_EDGESC + edge];
  if (lane < 4) {
    float ae = (lane == 0) ? part.x : (lane == 1) ? part.y : (lane == 2) ? part.z : part.w;
    float s = ws[WS_ASRC + (size_t)src * 4 + lane] +
              ws[WS_ADST + (size_t)dst * 4 + lane] + ae;
    s = (s >= 0.f) ? s : LRELU_ALPHA * s;
    ws[WS_SCORE + (size_t)edge * 4 + lane] = s;
    atomicMax(&smax[lane], fenc(s));
  }
  __syncthreads();
  if (tid < 4) atomicMax(&((unsigned int*)(ws + WS_RED))[tid], smax[tid]);
}

// ---------------------------------------------------------------------------
// 5) p = exp(s - max), global per-head sum   (thread per edge)
// ---------------------------------------------------------------------------
__global__ void __launch_bounds__(256)
k_edge_exp(float* __restrict__ ws) {
  __shared__ float ssum[4];
  int tid = threadIdx.x;
  if (tid < 4) ssum[tid] = 0.f;
  __syncthreads();

  size_t e = (size_t)blockIdx.x * 256 + tid;
  const unsigned int* gmax = (const unsigned int*)(ws + WS_RED);
  float m0 = fdec(gmax[0]), m1 = fdec(gmax[1]), m2 = fdec(gmax[2]), m3 = fdec(gmax[3]);
  float4* sc4 = (float4*)(ws + WS_SCORE);
  float4 s = sc4[e];
  float4 p;
  p.x = __expf(s.x - m0); p.y = __expf(s.y - m1);
  p.z = __expf(s.z - m2); p.w = __expf(s.w - m3);
  sc4[e] = p;

  float4 r = p;
  #pragma unroll
  for (int off = 16; off > 0; off >>= 1) {
    r.x += __shfl_xor(r.x, off); r.y += __shfl_xor(r.y, off);
    r.z += __shfl_xor(r.z, off); r.w += __shfl_xor(r.w, off);
  }
  if ((tid & 31) == 0) {
    atomicAdd(&ssum[0], r.x); atomicAdd(&ssum[1], r.y);
    atomicAdd(&ssum[2], r.z); atomicAdd(&ssum[3], r.w);
  }
  __syncthreads();
  if (tid < 4) atomicAdd(&ws[WS_RED + 4 + tid], ssum[tid]);
}

// ---------------------------------------------------------------------------
// 6) scatter: out[dst] += (p/sum) * Wh[src]   (wave per edge; Wh L2-resident)
// ---------------------------------------------------------------------------
__global__ void __launch_bounds__(256)
k_scatter(const int* __restrict__ ei, float* __restrict__ out,
          const float* __restrict__ ws) {
  int edge = blockIdx.x * 8 + (threadIdx.x >> 5);
  int lane = threadIdx.x & 31;
  int head = lane >> 3;
  int src = ei[edge], dst = ei[N_EDGESC + edge];
  float4 wv = ((const float4*)(ws + WS_WH + (size_t)src * IN_FC))[lane];
  float p   = ws[WS_SCORE + (size_t)edge * 4 + head];
  float sum = ws[WS_RED + 4 + head];
  float a = p / sum;
  float* o = out + (size_t)dst * IN_FC + lane * 4;
  unsafeAtomicAdd(o + 0, a * wv.x);
  unsafeAtomicAdd(o + 1, a * wv.y);
  unsafeAtomicAdd(o + 2, a * wv.z);
  unsafeAtomicAdd(o + 3, a * wv.w);
}

// ---------------------------------------------------------------------------
// 7) in-place ReLU on output
// ---------------------------------------------------------------------------
__global__ void k_relu(float* __restrict__ out) {
  size_t i = (size_t)blockIdx.x * blockDim.x + threadIdx.x;
  float4* o4 = (float4*)out;
  const size_t n4 = (size_t)N_NODESC * IN_FC / 4;
  for (size_t k = i; k < n4; k += (size_t)gridDim.x * blockDim.x) {
    float4 v = o4[k];
    v.x = fmaxf(v.x, 0.f); v.y = fmaxf(v.y, 0.f);
    v.z = fmaxf(v.z, 0.f); v.w = fmaxf(v.w, 0.f);
    o4[k] = v;
  }
}

extern "C" void kernel_launch(void* const* d_in, const int* in_sizes, int n_in,
                              void* d_out, int out_size, void* d_ws, size_t ws_size,
                              hipStream_t stream) {
  const int*   edge_index = (const int*)d_in[0];
  const float* h          = (const float*)d_in[1];
  const float* edge_feat  = (const float*)d_in[2];
  const float* W          = (const float*)d_in[3];
  const float* W_e        = (const float*)d_in[4];
  const float* a_src      = (const float*)d_in[5];
  const float* a_dst      = (const float*)d_in[6];
  const float* a_edge     = (const float*)d_in[7];
  float* out = (float*)d_out;
  float* ws  = (float*)d_ws;

  k_init      <<<2048, 256, 0, stream>>>(out, ws);
  k_wea       <<<1, 128, 0, stream>>>(W_e, a_edge, ws);
  k_gemm_wh   <<<391, 256, 0, stream>>>(h, W, ws);         // 3125 16-row tiles
  k_alpha_node<<<N_NODESC / 8, 256, 0, stream>>>(a_src, a_dst, ws);
  k_edge_score<<<N_EDGESC / 8, 256, 0, stream>>>(edge_index, edge_feat, ws);
  k_edge_exp  <<<N_EDGESC / 256, 256, 0, stream>>>(ws);
  k_scatter   <<<N_EDGESC / 8, 256, 0, stream>>>(edge_index, out, ws);
  k_relu      <<<2048, 256, 0, stream>>>(out);
}
